// ProposedModel_58428735095628
// MI455X (gfx1250) — matrix operand order repaired and checked by reading, not compiled, not run on hardware
//
#include <hip/hip_runtime.h>
#include <hip/hip_bf16.h>

// ---------------------------------------------------------------------------
// Problem constants (match reference: B=8, N=1024, D=1024, H=64, K=20)
// ---------------------------------------------------------------------------
constexpr int kB  = 8;
constexpr int kN  = 1024;
constexpr int kD  = 1024;
constexpr int kH  = 64;
constexpr int kK  = 20;

typedef __attribute__((ext_vector_type(16))) __bf16 v16bf;
typedef __attribute__((ext_vector_type(8)))  __bf16 v8bf;
typedef __attribute__((ext_vector_type(8)))  float  v8f;
typedef __attribute__((ext_vector_type(4)))  unsigned int v4u;
typedef __attribute__((ext_vector_type(8)))  unsigned int v8u;

// ---------------------------------------------------------------------------
// Tensor Data Mover: issue a 2D bf16 tile load Global -> LDS.
// D# layout per CDNA5 ISA:
//   group0: [1:0]=count=1, [63:32]=lds_addr, [120:64]=global_addr, [127:126]=2
//   group1: [17:16]=data_size(1 => 2 bytes), [79:48]=tensor_dim0,
//           [111:80]=tensor_dim1, [127:112]=tile_dim0, [143:128]=tile_dim1,
//           [207:160]=tensor_dim0_stride
// Tile rows land contiguously in LDS: LDS[r*tile_d0 + e].
// Uniform (per-wave scalar) arguments only; EXEC is ignored by TDM.
// ---------------------------------------------------------------------------
__device__ __forceinline__ void tdm_load_tile_bf16(const __bf16* gptr,
                                                   unsigned lds_off,
                                                   unsigned tile_d0, unsigned tile_d1,
                                                   unsigned tensor_d0, unsigned tensor_d1,
                                                   unsigned stride0) {
  unsigned long long ga = (unsigned long long)(uintptr_t)gptr;
  v4u g0;
  g0[0] = 1u;                                            // count=1, user descriptor
  g0[1] = lds_off;                                       // lds_addr (bytes)
  g0[2] = (unsigned)(ga & 0xffffffffu);                  // global_addr[31:0]
  g0[3] = (unsigned)((ga >> 32) & 0x01ffffffu) | (2u << 30);  // addr[56:32] | type=2
  v8u g1;
  g1[0] = (1u << 16);                                    // data_size=1 (2-byte elems)
  g1[1] = (tensor_d0 & 0xffffu) << 16;                   // tensor_dim0[15:0]
  g1[2] = ((tensor_d0 >> 16) & 0xffffu) | ((tensor_d1 & 0xffffu) << 16);
  g1[3] = ((tensor_d1 >> 16) & 0xffffu) | ((tile_d0 & 0xffffu) << 16);
  g1[4] = tile_d1 & 0xffffu;                             // tile_dim1 (tile_dim2=0)
  g1[5] = stride0;                                       // tensor_dim0_stride[31:0]
  g1[6] = 0u;                                            // stride0[47:32]=0, dim1_stride lo
  g1[7] = 0u;
  asm volatile("tensor_load_to_lds %0, %1" :: "s"(g0), "s"(g1) : "memory");
}

__device__ __forceinline__ unsigned lds_offset_of(const void* p) {
  // Generic LDS pointer: addr[31:0] is the LDS byte offset (aperture rules).
  return (unsigned)(uintptr_t)p;
}

// ---------------------------------------------------------------------------
// fp32 -> bf16 bulk convert
// ---------------------------------------------------------------------------
__global__ __launch_bounds__(256) void cvt_bf16_kernel(const float* __restrict__ in,
                                                       __bf16* __restrict__ out, int n) {
  int i = blockIdx.x * 256 + threadIdx.x;
  if (i < n) out[i] = (__bf16)in[i];
}

// ---------------------------------------------------------------------------
// Row squared-norms: one wave per row, lane-strided + shuffle reduction
// ---------------------------------------------------------------------------
__global__ __launch_bounds__(256) void rownorm_kernel(const float* __restrict__ f,
                                                      float* __restrict__ sq) {
  const int row  = blockIdx.x * 8 + (threadIdx.x >> 5);
  const int lane = threadIdx.x & 31;
  const float* r = f + (size_t)row * kD;
  float s = 0.f;
  for (int c = lane; c < kD; c += 32) { float v = r[c]; s += v * v; }
  #pragma unroll
  for (int o = 16; o > 0; o >>= 1) s += __shfl_xor(s, o, 32);
  if (lane == 0) sq[row] = s;
}

// ---------------------------------------------------------------------------
// Gram/distance kernel: dist[i][j] = |i|^2 + |j|^2 - 2 * f_i . f_j
// bf16 WMMA, f32 accumulate. Block = 8 waves; tile 128 rows x 64 cols.
// B-operand tiles (64 rows x 32 K-elems) are fed by TDM into double-buffered
// LDS; wave 0 drives the TDM pipeline, s_wait_tensorcnt retires tiles in order.
// ---------------------------------------------------------------------------
__global__ __launch_bounds__(256) void gram_dist_kernel(const __bf16* __restrict__ fb,
                                                        const float*  __restrict__ sq,
                                                        float* __restrict__ dist) {
  __shared__ __align__(16) __bf16 btile[2][64 * 32];
  const int lane = threadIdx.x & 31;
  const int wave = threadIdx.x >> 5;
  const int half = lane >> 4;
  const int l15  = lane & 15;
  const int r0   = blockIdx.y * 128 + wave * 16;
  const int c0   = blockIdx.x * 64;

  const unsigned lds0 = lds_offset_of(&btile[0][0]);
  const unsigned lds1 = lds_offset_of(&btile[1][0]);

  v8f acc[4];
  #pragma unroll
  for (int n = 0; n < 4; ++n)
    #pragma unroll
    for (int v = 0; v < 8; ++v) acc[n][v] = 0.f;

  // Prologue: stage K-chunk 0 of rows [c0, c0+64) into buffer 0.
  if (wave == 0)
    tdm_load_tile_bf16(fb + (size_t)c0 * kD, lds0, /*tile*/32, 64,
                       /*tensor*/kD, kN, /*stride*/kD);

  const int arow = r0 + l15;
  for (int kk = 0; kk < kD; kk += 32) {
    const int cur = (kk >> 5) & 1;
    __syncthreads();   // all waves done reading the buffer TDM is about to overwrite
    if (wave == 0) {
      if (kk + 32 < kD) {
        tdm_load_tile_bf16(fb + (size_t)c0 * kD + (kk + 32), cur ? lds0 : lds1,
                           32, 64, kD, kN, kD);
        __builtin_amdgcn_s_wait_tensorcnt(1);   // current tile complete
      } else {
        __builtin_amdgcn_s_wait_tensorcnt(0);
      }
    }
    __syncthreads();   // publish buf[cur] to all waves

    // A fragment: lane half 0 -> K {0..15}, half 1 -> K {16..31}
    v16bf a;
    {
      const v8bf x0 = *(const v8bf*)(fb + (size_t)arow * kD + kk + half * 16);
      const v8bf x1 = *(const v8bf*)(fb + (size_t)arow * kD + kk + half * 16 + 8);
      #pragma unroll
      for (int e = 0; e < 8; ++e) { a[e] = x0[e]; a[8 + e] = x1[e]; }
    }
    const __bf16* bbase = &btile[cur][0];
    #pragma unroll
    for (int n = 0; n < 4; ++n) {
      const int r = n * 16 + l15;     // tile-local B row (global col c0 + r)
      const v16bf bfrag = *(const v16bf*)(bbase + r * 32 + half * 16);
      acc[n] = __builtin_amdgcn_wmma_f32_16x16x32_bf16(
          false, a, false, bfrag, (short)0, acc[n], false, false);
    }
  }

  #pragma unroll
  for (int n = 0; n < 4; ++n) {
    const int col = c0 + n * 16 + l15;
    const float sqj = sq[col];
    #pragma unroll
    for (int v = 0; v < 8; ++v) {
      const int row = r0 + v + 8 * half;                 // C/D layout: M = v + 8*half
      dist[(size_t)row * kN + col] = sq[row] + sqj - 2.0f * acc[n][v];
    }
  }
}

// ---------------------------------------------------------------------------
// Per-row top-k (k=20 smallest, self excluded): one thread per row,
// register insertion list.
// ---------------------------------------------------------------------------
__global__ __launch_bounds__(256) void topk_kernel(const float* __restrict__ dist,
                                                   int* __restrict__ idx) {
  const int i = blockIdx.x * 256 + threadIdx.x;
  if (i >= kN) return;
  float bd[kK]; int bi[kK];
  #pragma unroll
  for (int t = 0; t < kK; ++t) { bd[t] = 3.4e38f; bi[t] = 0; }
  const float* row = dist + (size_t)i * kN;
  for (int j = 0; j < kN; ++j) {
    if (j == i) continue;
    const float d = row[j];
    if (d < bd[kK - 1]) {
      int t = kK - 1;
      while (t > 0 && bd[t - 1] > d) { bd[t] = bd[t - 1]; bi[t] = bi[t - 1]; --t; }
      bd[t] = d; bi[t] = j;
    }
  }
  for (int t = 0; t < kK; ++t) idx[i * kK + t] = bi[t];
}

// ---------------------------------------------------------------------------
// GIN aggregation, layer 1 (width D): h_bf[i,:] = bf16( f[i,:] + sum_nb f[j,:] )
// ---------------------------------------------------------------------------
__global__ __launch_bounds__(256) void agg_wide_kernel(const float* __restrict__ f,
                                                       const int* __restrict__ idx,
                                                       __bf16* __restrict__ hb) {
  __shared__ int nb[kK];
  const int i = blockIdx.x;
  if (threadIdx.x < kK) nb[threadIdx.x] = idx[i * kK + threadIdx.x];
  __syncthreads();
  for (int c = threadIdx.x; c < kD; c += 256) {
    float s = f[(size_t)i * kD + c];
    #pragma unroll
    for (int t = 0; t < kK; ++t) s += f[(size_t)nb[t] * kD + c];
    hb[(size_t)i * kD + c] = (__bf16)s;
  }
}

// GIN aggregation, layer 2 (width H=64)
__global__ __launch_bounds__(64) void agg_narrow_kernel(const float* __restrict__ z,
                                                        const int* __restrict__ idx,
                                                        __bf16* __restrict__ hb) {
  __shared__ int nb[kK];
  const int i = blockIdx.x;
  const int c = threadIdx.x;
  if (threadIdx.x < kK) nb[threadIdx.x] = idx[i * kK + threadIdx.x];
  __syncthreads();
  float s = z[(size_t)i * kH + c];
  #pragma unroll
  for (int t = 0; t < kK; ++t) s += z[(size_t)nb[t] * kH + c];
  hb[(size_t)i * kH + c] = (__bf16)s;
}

// ---------------------------------------------------------------------------
// Generic MLP GEMM: out = act(A[M,Kd] @ Bw[Kd,64] + bias), bf16 WMMA.
// The 32x64 bf16 weight k-slice is TDM-fed into double-buffered LDS.
// Epilogue optionally writes f32 and/or bf16.
// ---------------------------------------------------------------------------
__global__ __launch_bounds__(256) void mlp_gemm_kernel(const __bf16* __restrict__ A,
                                                       const __bf16* __restrict__ Bw,
                                                       const float*  __restrict__ bias,
                                                       float*  __restrict__ outF,
                                                       __bf16* __restrict__ outB,
                                                       int Kd, int relu) {
  __shared__ __align__(16) __bf16 bs[2][32 * 64];
  const int lane = threadIdx.x & 31;
  const int wave = threadIdx.x >> 5;
  const int half = lane >> 4;
  const int l15  = lane & 15;
  const int r0   = blockIdx.x * 128 + wave * 16;
  const int arow = r0 + l15;

  const unsigned lds0 = lds_offset_of(&bs[0][0]);
  const unsigned lds1 = lds_offset_of(&bs[1][0]);

  v8f acc[4];
  #pragma unroll
  for (int n = 0; n < 4; ++n)
    #pragma unroll
    for (int v = 0; v < 8; ++v) acc[n][v] = 0.f;

  // Prologue: weight rows [0,32) x 64 cols (contiguous, stride 64).
  if (wave == 0)
    tdm_load_tile_bf16(Bw, lds0, /*tile*/64, 32, /*tensor*/kH, (unsigned)Kd, /*stride*/kH);

  for (int kk = 0; kk < Kd; kk += 32) {
    const int cur = (kk >> 5) & 1;
    __syncthreads();
    if (wave == 0) {
      if (kk + 32 < Kd) {
        tdm_load_tile_bf16(Bw + (size_t)(kk + 32) * kH, cur ? lds0 : lds1,
                           64, 32, kH, (unsigned)Kd, kH);
        __builtin_amdgcn_s_wait_tensorcnt(1);
      } else {
        __builtin_amdgcn_s_wait_tensorcnt(0);
      }
    }
    __syncthreads();

    v16bf a;
    {
      const v8bf x0 = *(const v8bf*)(A + (size_t)arow * Kd + kk + half * 16);
      const v8bf x1 = *(const v8bf*)(A + (size_t)arow * Kd + kk + half * 16 + 8);
      #pragma unroll
      for (int e = 0; e < 8; ++e) { a[e] = x0[e]; a[8 + e] = x1[e]; }
    }
    const __bf16* bbase = &bs[cur][0];
    #pragma unroll
    for (int n = 0; n < 4; ++n) {
      const int col = n * 16 + l15;
      v16bf bfrag;
      #pragma unroll
      for (int e = 0; e < 16; ++e) bfrag[e] = bbase[(half * 16 + e) * 64 + col];
      acc[n] = __builtin_amdgcn_wmma_f32_16x16x32_bf16(
          false, a, false, bfrag, (short)0, acc[n], false, false);
    }
  }

  #pragma unroll
  for (int n = 0; n < 4; ++n) {
    const int col = n * 16 + l15;
    const float bv = bias[col];
    #pragma unroll
    for (int v = 0; v < 8; ++v) {
      const int row = r0 + v + 8 * half;
      float val = acc[n][v] + bv;
      if (relu) val = fmaxf(val, 0.f);
      if (outF) outF[(size_t)row * kH + col] = val;
      if (outB) outB[(size_t)row * kH + col] = (__bf16)val;
    }
  }
}

// ---------------------------------------------------------------------------
// sim = z1[1024,64] @ z2^T -> out[1024,1024] (f32), bf16 WMMA, K=64
// ---------------------------------------------------------------------------
__global__ __launch_bounds__(256) void sim_gemm_kernel(const __bf16* __restrict__ z1,
                                                       const __bf16* __restrict__ z2,
                                                       float* __restrict__ out) {
  const int lane = threadIdx.x & 31;
  const int wave = threadIdx.x >> 5;
  const int half = lane >> 4;
  const int l15  = lane & 15;
  const int r0   = blockIdx.y * 128 + wave * 16;
  const int c0   = blockIdx.x * 64;

  v8f acc[4];
  #pragma unroll
  for (int n = 0; n < 4; ++n)
    #pragma unroll
    for (int v = 0; v < 8; ++v) acc[n][v] = 0.f;

  const int arow = r0 + l15;
  #pragma unroll
  for (int kk = 0; kk < kH; kk += 32) {
    v16bf a;
    const v8bf x0 = *(const v8bf*)(z1 + (size_t)arow * kH + kk + half * 16);
    const v8bf x1 = *(const v8bf*)(z1 + (size_t)arow * kH + kk + half * 16 + 8);
    #pragma unroll
    for (int e = 0; e < 8; ++e) { a[e] = x0[e]; a[8 + e] = x1[e]; }
    #pragma unroll
    for (int n = 0; n < 4; ++n) {
      const int col = c0 + n * 16 + l15;
      const v16bf bfrag = *(const v16bf*)(z2 + (size_t)col * kH + kk + half * 16);
      acc[n] = __builtin_amdgcn_wmma_f32_16x16x32_bf16(
          false, a, false, bfrag, (short)0, acc[n], false, false);
    }
  }
  #pragma unroll
  for (int n = 0; n < 4; ++n) {
    const int col = c0 + n * 16 + l15;
    #pragma unroll
    for (int v = 0; v < 8; ++v)
      out[(size_t)(r0 + v + 8 * half) * kN + col] = acc[n][v];
  }
}

// ---------------------------------------------------------------------------
// Row softmax in-place on [kN, kN]; one 256-thread block per row.
// ---------------------------------------------------------------------------
__global__ __launch_bounds__(256) void softmax_kernel(float* __restrict__ out) {
  const int i = blockIdx.x;
  float* row = out + (size_t)i * kN;
  __shared__ float redm[8];
  __shared__ float reds[8];
  const int lane = threadIdx.x & 31, wave = threadIdx.x >> 5;

  float m = -3.402823e38f;
  for (int c = threadIdx.x; c < kN; c += 256) m = fmaxf(m, row[c]);
  #pragma unroll
  for (int o = 16; o > 0; o >>= 1) m = fmaxf(m, __shfl_xor(m, o, 32));
  if (lane == 0) redm[wave] = m;
  __syncthreads();
  m = redm[0];
  #pragma unroll
  for (int w = 1; w < 8; ++w) m = fmaxf(m, redm[w]);

  float s = 0.f;
  for (int c = threadIdx.x; c < kN; c += 256) {
    float e = __expf(row[c] - m);
    row[c] = e;
    s += e;
  }
  #pragma unroll
  for (int o = 16; o > 0; o >>= 1) s += __shfl_xor(s, o, 32);
  if (lane == 0) reds[wave] = s;
  __syncthreads();
  s = 0.f;
  #pragma unroll
  for (int w = 0; w < 8; ++w) s += reds[w];
  const float inv = 1.0f / s;
  for (int c = threadIdx.x; c < kN; c += 256) row[c] *= inv;
}

// ---------------------------------------------------------------------------
// Host orchestration
// ---------------------------------------------------------------------------
extern "C" void kernel_launch(void* const* d_in, const int* in_sizes, int n_in,
                              void* d_out, int out_size, void* d_ws, size_t ws_size,
                              hipStream_t stream) {
  (void)in_sizes; (void)n_in; (void)out_size; (void)ws_size;

  const float* f1  = (const float*)d_in[0];
  const float* f2  = (const float*)d_in[1];
  const float* W1a = (const float*)d_in[2];
  const float* b1a = (const float*)d_in[3];
  const float* W2a = (const float*)d_in[4];
  const float* b2a = (const float*)d_in[5];
  const float* W1b = (const float*)d_in[6];
  const float* b1b = (const float*)d_in[7];
  const float* W2b = (const float*)d_in[8];
  const float* b2b = (const float*)d_in[9];
  // d_in[10] = k (compile-time 20)

  char* ws = (char*)d_ws;
  size_t off = 0;
  auto take = [&](size_t bytes) -> char* {
    char* p = ws + off;
    off += (bytes + 255) & ~(size_t)255;
    return p;
  };

  __bf16* fb_all = (__bf16*)take((size_t)2 * kB * kN * kD * sizeof(__bf16));
  __bf16* w1a_b  = (__bf16*)take((size_t)kD * kH * sizeof(__bf16));
  __bf16* w2a_b  = (__bf16*)take((size_t)kH * kH * sizeof(__bf16));
  __bf16* w1b_b  = (__bf16*)take((size_t)kH * kH * sizeof(__bf16));
  __bf16* w2b_b  = (__bf16*)take((size_t)kH * kH * sizeof(__bf16));
  float*  sq_all = (float*) take((size_t)2 * kB * kN * sizeof(float));
  float*  dist   = (float*) take((size_t)kN * kN * sizeof(float));
  int*    idxb   = (int*)   take((size_t)kN * kK * sizeof(int));
  __bf16* h_bf   = (__bf16*)take((size_t)kN * kD * sizeof(__bf16));
  __bf16* t_bf   = (__bf16*)take((size_t)kN * kH * sizeof(__bf16));
  float*  z_f    = (float*) take((size_t)kN * kH * sizeof(float));
  __bf16* h2_bf  = (__bf16*)take((size_t)kN * kH * sizeof(__bf16));
  __bf16* t2_bf  = (__bf16*)take((size_t)kN * kH * sizeof(__bf16));
  __bf16* zfin   = (__bf16*)take((size_t)2 * kB * kN * kH * sizeof(__bf16));

  auto cvt = [&](const float* src, __bf16* dst, int n) {
    cvt_bf16_kernel<<<(n + 255) / 256, 256, 0, stream>>>(src, dst, n);
  };
  cvt(f1, fb_all, kB * kN * kD);
  cvt(f2, fb_all + (size_t)kB * kN * kD, kB * kN * kD);
  cvt(W1a, w1a_b, kD * kH);
  cvt(W2a, w2a_b, kH * kH);
  cvt(W1b, w1b_b, kH * kH);
  cvt(W2b, w2b_b, kH * kH);

  for (int which = 0; which < 2; ++which) {
    for (int b = 0; b < kB; ++b) {
      const int s = which * kB + b;
      const float*  f  = ((which == 0) ? f1 : f2) + (size_t)b * kN * kD;
      const __bf16* fb = fb_all + (size_t)s * kN * kD;
      float* sq = sq_all + (size_t)s * kN;

      rownorm_kernel<<<kN / 8, 256, 0, stream>>>(f, sq);
      gram_dist_kernel<<<dim3(kN / 64, kN / 128), 256, 0, stream>>>(fb, sq, dist);
      topk_kernel<<<kN / 256, 256, 0, stream>>>(dist, idxb);

      // GIN layer 1 (width D -> H)
      agg_wide_kernel<<<kN, 256, 0, stream>>>(f, idxb, h_bf);
      mlp_gemm_kernel<<<kN / 128, 256, 0, stream>>>(h_bf, w1a_b, b1a, nullptr, t_bf, kD, 1);
      mlp_gemm_kernel<<<kN / 128, 256, 0, stream>>>(t_bf, w2a_b, b2a, z_f, nullptr, kH, 0);

      // GIN layer 2 (width H -> H)
      agg_narrow_kernel<<<kN, 64, 0, stream>>>(z_f, idxb, h2_bf);
      mlp_gemm_kernel<<<kN / 128, 256, 0, stream>>>(h2_bf, w1b_b, b1b, nullptr, t2_bf, kH, 1);
      mlp_gemm_kernel<<<kN / 128, 256, 0, stream>>>(t2_bf, w2b_b, b2b, nullptr,
                                                    zfin + (size_t)s * kN * kH, kH, 0);
    }
  }

  for (int b = 0; b < kB; ++b) {
    float* ob = (float*)d_out + (size_t)b * kN * kN;
    sim_gemm_kernel<<<dim3(kN / 64, kN / 128), 256, 0, stream>>>(
        zfin + (size_t)b * kN * kH, zfin + (size_t)(kB + b) * kN * kH, ob);
    softmax_kernel<<<kN, 256, 0, stream>>>(ob);
  }
}
